// VAE_Encoder_64372969832520
// MI455X (gfx1250) — matrix operand order repaired and checked by reading, not compiled, not run on hardware
//
#include <hip/hip_runtime.h>
#include <hip/hip_bf16.h>

typedef _Float16 f16;
typedef __attribute__((ext_vector_type(16))) _Float16 v16h;
typedef __attribute__((ext_vector_type(8)))  _Float16 v8h;
typedef __attribute__((ext_vector_type(2)))  _Float16 v2h;
typedef __attribute__((ext_vector_type(8)))  float    v8f;

#define BB   64
#define TT   1024
#define HH   12
#define LL   120
#define KCONV 5
#define T1   1020
#define T2   1016
#define G4   48      // 4*H
#define RING 16      // ring depth per layer boundary (power of 2)

#define LOG2E 1.4426950408889634f

// Branch-free fast activations: v_exp_f32 (TRANS, co-executes) + v_rcp_f32.
__device__ __forceinline__ float sigf(float x) {
    return __builtin_amdgcn_rcpf(1.0f + __builtin_amdgcn_exp2f(-LOG2E * x));
}
__device__ __forceinline__ float tanhfast(float x) {
    // tanh(x) = 1 - 2/(1+e^{2x}); exp2 overflow->inf->rcp->0 gives +/-1 tails.
    return 1.0f - 2.0f * __builtin_amdgcn_rcpf(1.0f + __builtin_amdgcn_exp2f(2.0f * LOG2E * x));
}

// ---------------- conv1 (valid, K=5): x(64,6,1024) -> y1(64,6,1020) ----------------
__global__ void k_conv1(const float* __restrict__ x, const float* __restrict__ w,
                        const float* __restrict__ b, float* __restrict__ y1) {
    int id = blockIdx.x * blockDim.x + threadIdx.x;
    if (id >= BB * 6 * T1) return;
    int t = id % T1, c = (id / T1) % 6, bb = id / (6 * T1);
    float acc = b[c];
#pragma unroll
    for (int ci = 0; ci < 6; ++ci)
#pragma unroll
        for (int k = 0; k < KCONV; ++k)
            acc += x[bb * 6 * TT + ci * TT + t + k] * w[c * 30 + ci * KCONV + k];
    y1[id] = acc;
}

// ---------------- per-channel batch stats: mean + rsqrt(var+eps) ----------------
__global__ void k_stats(const float* __restrict__ y, int tlen, float* __restrict__ stats) {
    __shared__ float ss[256], sq[256];
    int c = blockIdx.x;
    float s = 0.f, q = 0.f;
    for (int i = threadIdx.x; i < BB * tlen; i += 256) {
        float v = y[(i / tlen) * 6 * tlen + c * tlen + (i % tlen)];
        s += v; q += v * v;
    }
    ss[threadIdx.x] = s; sq[threadIdx.x] = q;
    __syncthreads();
    for (int o = 128; o; o >>= 1) {
        if (threadIdx.x < o) { ss[threadIdx.x] += ss[threadIdx.x + o]; sq[threadIdx.x] += sq[threadIdx.x + o]; }
        __syncthreads();
    }
    if (threadIdx.x == 0) {
        float n = (float)(BB * tlen);
        float m = ss[0] / n, v = sq[0] / n - m * m;
        stats[c] = m;
        stats[8 + c] = rsqrtf(v + 1e-5f);
    }
}

// ---------------- bn1+relu fused into conv2: y1 -> y2(64,6,1016) ----------------
__global__ void k_conv2(const float* __restrict__ y1, const float* __restrict__ st,
                        const float* __restrict__ g, const float* __restrict__ be,
                        const float* __restrict__ w, const float* __restrict__ b,
                        float* __restrict__ y2) {
    int id = blockIdx.x * blockDim.x + threadIdx.x;
    if (id >= BB * 6 * T2) return;
    int t = id % T2, c = (id / T2) % 6, bb = id / (6 * T2);
    float acc = b[c];
#pragma unroll
    for (int ci = 0; ci < 6; ++ci) {
        float mu = st[ci], inv = st[8 + ci], ga = g[ci], bt = be[ci];
#pragma unroll
        for (int k = 0; k < KCONV; ++k) {
            float v = y1[bb * 6 * T1 + ci * T1 + t + k];
            v = fmaxf((v - mu) * inv * ga + bt, 0.f);
            acc += v * w[c * 30 + ci * KCONV + k];
        }
    }
    y2[id] = acc;
}

// ---------------- bn2+relu + attention proj -> feat f16 [t][b][h] ----------------
__global__ void k_feat(const float* __restrict__ y2, const float* __restrict__ st,
                       const float* __restrict__ g, const float* __restrict__ be,
                       const float* __restrict__ aw, const float* __restrict__ ab,
                       f16* __restrict__ feat) {
    int id = blockIdx.x * blockDim.x + threadIdx.x;
    if (id >= T2 * BB * HH) return;
    int h = id % HH, bb = (id / HH) % BB, t = id / (BB * HH);
    float acc = ab[h];
#pragma unroll
    for (int c = 0; c < 6; ++c) {
        float v = y2[bb * 6 * T2 + c * T2 + t];
        v = fmaxf((v - st[c]) * st[8 + c] * g[c] + be[c], 0.f);
        acc += aw[h * 6 + c] * v;
    }
    feat[(size_t)t * (BB * HH) + bb * HH + h] = (f16)fmaxf(acc, 0.f);
}

// ---------------- pack combined LSTM weights: Wcomb[l][32][48] f16, bias[l][48] ----------------
__global__ void k_wprep(const float* __restrict__ wih, const float* __restrict__ whh,
                        const float* __restrict__ bih, const float* __restrict__ bhh,
                        f16* __restrict__ wcomb, float* __restrict__ bcomb) {
    int id = blockIdx.x * blockDim.x + threadIdx.x;
    if (id < LL * G4) bcomb[id] = bih[id] + bhh[id];
    if (id >= LL * 32 * G4) return;
    int n = id % G4, k = (id / G4) % 32, l = id / (32 * G4);
    float v = 0.f;
    if (k < 12)      v = wih[l * 576 + n * 12 + k];
    else if (k < 24) v = whh[l * 576 + n * 12 + (k - 12)];
    wcomb[id] = (f16)v;
}

__global__ void k_flags(int* __restrict__ flags) { flags[threadIdx.x] = 0; }

// ---------------- wavefront-pipelined 120-layer LSTM, WMMA gates ----------------
__global__ void __launch_bounds__(128) k_lstm(const f16* __restrict__ feat,
                                              const f16* __restrict__ wcomb,
                                              const float* __restrict__ bcomb,
                                              f16* __restrict__ ring,
                                              int* __restrict__ flags,
                                              float* __restrict__ hlast) {
    __shared__ f16   Ap[64 * 32];   // [batch row][K]: 0..11 = x_t, 12..23 = h_{t-1}, 24..31 = 0
    __shared__ float gt[64 * 48];   // gate outputs
    __shared__ float cs[64 * 12];   // cell state

    const int li   = blockIdx.x;        // layer
    const int tid  = threadIdx.x;
    const int lane = tid & 31;
    const int wv   = tid >> 5;
    const int m0   = wv * 16;           // wave owns batch rows m0..m0+15
    int* produced = flags;
    int* consumed = flags + 128;

    for (int i = tid; i < 64 * 32; i += 128) Ap[i] = (f16)0.f;
    for (int i = tid; i < 64 * 12; i += 128) cs[i] = 0.f;

    // B fragments (32x16 f16 per N tile) + bias, resident in registers for the whole layer.
    // Layout: lanes 0-15 hold K=0..7 & 16..23 of column N=lane; lanes 16-31 hold K=8..15 & 24..31 of N=lane-16.
    v16h bf0, bf1, bf2;
    float cb[3];
    {
        int n  = lane & 15;
        int kb = (lane < 16) ? 0 : 8;
        const f16* wl = wcomb + (size_t)li * 32 * G4;
#pragma unroll
        for (int e = 0; e < 16; ++e) {
            bf0[e] = wl[(kb + (e & 7) + ((e >> 3) << 4)) * G4 + 0  + n];
            bf1[e] = wl[(kb + (e & 7) + ((e >> 3) << 4)) * G4 + 16 + n];
            bf2[e] = wl[(kb + (e & 7) + ((e >> 3) << 4)) * G4 + 32 + n];
        }
        cb[0] = bcomb[li * G4 + 0  + (lane & 15)];
        cb[1] = bcomb[li * G4 + 16 + (lane & 15)];
        cb[2] = bcomb[li * G4 + 32 + (lane & 15)];
    }
    __syncthreads();

    // A fragment addressing (16-bit A 16x32): lane<16 -> K 0..7 & 16..23 of row lane;
    // lane>=16 -> K 8..15 & 24..31 of row lane-16.
    const int arow = m0 + (lane & 15);
    const int akb  = (lane < 16) ? 0 : 8;

    // Elementwise mapping: lane's 6 items span exactly half a 12-wide row -> fixed row.
    const int brow = m0 + (lane >> 1);        // batch row for this lane's activations
    const int j0   = (lane & 1) * 6;          // hidden offset 0 or 6

    for (int t = 0; t < T2; ++t) {
        // ---- consume x_t from previous layer (or feat for layer 0) ----
        if (li > 0) {
            if (tid == 0)
                while (__hip_atomic_load(&produced[li], __ATOMIC_ACQUIRE, __HIP_MEMORY_SCOPE_AGENT) <= t)
                    __builtin_amdgcn_s_sleep(1);
            __syncthreads();
        }
        const f16* src = (li == 0) ? (feat + (size_t)t * 768)
                                   : (ring + ((size_t)li * RING + (t & (RING - 1))) * 768);
        // 768 halves = 384 dwords; 3 dword copies per thread into strided A panel.
        {
            const unsigned int* su = (const unsigned int*)src;
#pragma unroll
            for (int q = 0; q < 3; ++q) {
                int d = tid * 3 + q;                       // dword index 0..383
                int row = d / 6, c2 = (d % 6) * 2;         // 6 dwords per 12-col row
                *(unsigned int*)&Ap[row * 32 + c2] = su[d];
            }
        }
        __syncthreads();
        if (li > 0 && tid == 32)
            __hip_atomic_store(&consumed[li], t + 1, __ATOMIC_RELEASE, __HIP_MEMORY_SCOPE_AGENT);

        // ---- A fragment: two 16B LDS loads per lane ----
        v8h lo = *(const v8h*)&Ap[arow * 32 + akb];
        v8h hi = *(const v8h*)&Ap[arow * 32 + akb + 16];
        v16h a;
#pragma unroll
        for (int i = 0; i < 8; ++i) { a[i] = lo[i]; a[i + 8] = hi[i]; }

        // ---- 3x v_wmma_f32_16x16x32_f16: gates = [x|h] @ [Wih^T;Whh^T] + bias ----
#pragma unroll
        for (int nt = 0; nt < 3; ++nt) {
            v8f c;
#pragma unroll
            for (int r = 0; r < 8; ++r) c[r] = cb[nt];
            v16h bsel = (nt == 0) ? bf0 : (nt == 1) ? bf1 : bf2;
            c = __builtin_amdgcn_wmma_f32_16x16x32_f16(false, a, false, bsel,
                                                       (short)0, c, false, false);
            int gcol = nt * 16 + (lane & 15);
            int gro  = m0 + ((lane < 16) ? 0 : 8);
#pragma unroll
            for (int r = 0; r < 8; ++r) gt[(gro + r) * G4 + gcol] = c[r];
        }

        // ---- elementwise LSTM update: 6 hidden units of one batch row per lane ----
        float hq[6];
#pragma unroll
        for (int q = 0; q < 6; ++q) {
            int j    = j0 + q;
            float iv = gt[brow * G4 + j];
            float fv = gt[brow * G4 + 12 + j];
            float gv = gt[brow * G4 + 24 + j];
            float ov = gt[brow * G4 + 36 + j];
            float co = cs[brow * 12 + j];
            float cn = sigf(fv) * co + sigf(iv) * tanhfast(gv);
            float hn = sigf(ov) * tanhfast(cn);
            cs[brow * 12 + j] = cn;
            hq[q] = hn;
            if (t == T2 - 1) hlast[brow * (LL * HH) + li * HH + j] = hn;
        }
        // h_t back into A panel: 3 packed b32 LDS stores (row-local, 4B aligned).
#pragma unroll
        for (int q = 0; q < 3; ++q) {
            v2h p; p[0] = (f16)hq[2 * q]; p[1] = (f16)hq[2 * q + 1];
            *(v2h*)&Ap[brow * 32 + 12 + j0 + 2 * q] = p;
        }

        // ---- produce h_t for next layer via L2 ring + release flag ----
        if (li < LL - 1) {
            if (t >= RING && tid == 0)
                while (__hip_atomic_load(&consumed[li + 1], __ATOMIC_ACQUIRE, __HIP_MEMORY_SCOPE_AGENT)
                       < t - RING + 1)
                    __builtin_amdgcn_s_sleep(1);
            __syncthreads();
            // Each lane's 6 halves are contiguous at m0*12 + lane*6: 3 packed b32 stores.
            f16* dst = ring + ((size_t)(li + 1) * RING + (t & (RING - 1))) * 768 + (m0 * 12 + lane * 6);
#pragma unroll
            for (int q = 0; q < 3; ++q) {
                v2h p; p[0] = (f16)hq[2 * q]; p[1] = (f16)hq[2 * q + 1];
                *(v2h*)&dst[2 * q] = p;
            }
            __threadfence();
            __syncthreads();
            if (tid == 0)
                __hip_atomic_store(&produced[li + 1], t + 1, __ATOMIC_RELEASE, __HIP_MEMORY_SCOPE_AGENT);
        } else {
            __syncthreads();
        }
    }
}

// ---------------- head: lin1+relu, lin2+relu, mu, softplus(sigma) ----------------
__global__ void k_head(const float* __restrict__ hlast,
                       const float* __restrict__ l1w, const float* __restrict__ l1b,
                       const float* __restrict__ l2w, const float* __restrict__ l2b,
                       const float* __restrict__ muw, const float* __restrict__ mub,
                       const float* __restrict__ sw,  const float* __restrict__ sb,
                       float* __restrict__ out) {
    int m = blockIdx.x * blockDim.x + threadIdx.x;   // b*120 + l
    if (m >= BB * LL) return;
    float h[12], z1[12], z2[12];
#pragma unroll
    for (int k = 0; k < 12; ++k) h[k] = hlast[m * 12 + k];
#pragma unroll
    for (int j = 0; j < 12; ++j) {
        float a = l1b[j];
#pragma unroll
        for (int k = 0; k < 12; ++k) a += h[k] * l1w[j * 12 + k];
        z1[j] = fmaxf(a, 0.f);
    }
#pragma unroll
    for (int j = 0; j < 12; ++j) {
        float a = l2b[j];
#pragma unroll
        for (int k = 0; k < 12; ++k) a += z1[k] * l2w[j * 12 + k];
        z2[j] = fmaxf(a, 0.f);
        out[m * 12 + j] = z2[j];
    }
#pragma unroll
    for (int j = 0; j < 12; ++j) {
        float mu = mub[j], sg = sb[j];
#pragma unroll
        for (int k = 0; k < 12; ++k) { mu += z2[k] * muw[j * 12 + k]; sg += z2[k] * sw[j * 12 + k]; }
        out[BB * LL * 12 + m * 12 + j] = mu;
        // softplus via v_exp_f32 / v_log_f32, branch-free with large-x select
        float e  = __builtin_amdgcn_exp2f(LOG2E * sg);
        float sp = 0.69314718056f * __builtin_amdgcn_logf(1.0f + e);
        out[2 * BB * LL * 12 + m * 12 + j] = (sg > 20.f) ? sg : sp;
    }
}

static inline size_t alignup(size_t v) { return (v + 255) & ~(size_t)255; }

extern "C" void kernel_launch(void* const* d_in, const int* in_sizes, int n_in,
                              void* d_out, int out_size, void* d_ws, size_t ws_size,
                              hipStream_t stream) {
    const float* x    = (const float*)d_in[0];
    const float* w1   = (const float*)d_in[1];
    const float* b1   = (const float*)d_in[2];
    const float* g1   = (const float*)d_in[3];
    const float* be1  = (const float*)d_in[4];
    const float* w2   = (const float*)d_in[5];
    const float* b2   = (const float*)d_in[6];
    const float* g2   = (const float*)d_in[7];
    const float* be2  = (const float*)d_in[8];
    const float* aw   = (const float*)d_in[9];
    const float* ab   = (const float*)d_in[10];
    const float* wih  = (const float*)d_in[11];
    const float* whh  = (const float*)d_in[12];
    const float* bih  = (const float*)d_in[13];
    const float* bhh  = (const float*)d_in[14];
    const float* l1w  = (const float*)d_in[15];
    const float* l1b  = (const float*)d_in[16];
    const float* l2w  = (const float*)d_in[17];
    const float* l2b  = (const float*)d_in[18];
    const float* muw  = (const float*)d_in[19];
    const float* mub  = (const float*)d_in[20];
    const float* sw   = (const float*)d_in[21];
    const float* sb   = (const float*)d_in[22];
    float* out = (float*)d_out;

    char* p = (char*)d_ws;
    float* y1     = (float*)p; p += alignup((size_t)BB * 6 * T1 * 4);
    float* stats1 = (float*)p; p += alignup(16 * 4);
    float* y2     = (float*)p; p += alignup((size_t)BB * 6 * T2 * 4);
    float* stats2 = (float*)p; p += alignup(16 * 4);
    f16*   feat   = (f16*)p;   p += alignup((size_t)T2 * BB * HH * 2);
    f16*   wcomb  = (f16*)p;   p += alignup((size_t)LL * 32 * G4 * 2);
    float* bcomb  = (float*)p; p += alignup((size_t)LL * G4 * 4);
    f16*   ring   = (f16*)p;   p += alignup((size_t)LL * RING * BB * HH * 2);
    int*   flags  = (int*)p;   p += alignup(256 * 4);
    float* hlast  = (float*)p; p += alignup((size_t)BB * LL * HH * 4);

    k_flags<<<1, 256, 0, stream>>>(flags);
    k_conv1<<<(BB * 6 * T1 + 255) / 256, 256, 0, stream>>>(x, w1, b1, y1);
    k_stats<<<6, 256, 0, stream>>>(y1, T1, stats1);
    k_conv2<<<(BB * 6 * T2 + 255) / 256, 256, 0, stream>>>(y1, stats1, g1, be1, w2, b2, y2);
    k_stats<<<6, 256, 0, stream>>>(y2, T2, stats2);
    k_feat<<<(T2 * BB * HH + 255) / 256, 256, 0, stream>>>(y2, stats2, g2, be2, aw, ab, feat);
    k_wprep<<<(LL * 32 * G4 + 255) / 256, 256, 0, stream>>>(wih, whh, bih, bhh, wcomb, bcomb);
    k_lstm<<<LL, 128, 0, stream>>>(feat, wcomb, bcomb, ring, flags, hlast);
    k_head<<<(BB * LL + 255) / 256, 256, 0, stream>>>(hlast, l1w, l1b, l2w, l2b, muw, mub, sw, sb, out);
}
